// BaselineModel_54503134986367
// MI455X (gfx1250) — matrix operand order repaired and checked by reading, not compiled, not run on hardware
//
#include <hip/hip_runtime.h>
#include <hip/hip_bf16.h>
#include <math.h>

typedef __attribute__((ext_vector_type(16))) _Float16 v16h;
typedef __attribute__((ext_vector_type(8)))  _Float16 v8h;
typedef __attribute__((ext_vector_type(8)))  float    v8f;

#define WMMA_F16(a, b, c) \
  __builtin_amdgcn_wmma_f32_16x16x32_f16(false, (a), false, (b), (short)0, (c), false, false)

static constexpr int BB = 4096;   // batch
static constexpr int NN = 16;     // objects
static constexpr int DV = 256;    // visual dim
static constexpr int DL = 1024;   // instruction dim

// ---------------------------------------------------------------------------
// WMMA fragment helpers (ISA 7.12.2 layouts, wave32)
// ---------------------------------------------------------------------------

// A fragment: 16x32 f16 tile, row-major source with leading dim lda.
// lanes 0-15: M=lane, K in {0..7, 16..23}; lanes 16-31: M=lane-16, K in {8..15, 24..31}
// -> per lane: two contiguous 8-half runs at (row,kb) and (row,kb+16), kb = 0/8.
__device__ inline v16h load_a_f16(const _Float16* A, int lda, int lane) {
  const _Float16* p = A + (size_t)(lane & 15) * lda + ((lane >> 4) << 3);
  v8h lo = *(const v8h*)p;
  v8h hi = *(const v8h*)(p + 16);
  v16h a;
#pragma unroll
  for (int j = 0; j < 8; ++j) { a[j] = lo[j]; a[8 + j] = hi[j]; }
  return a;
}

// Packed B tile: 512 halves per (ktile, coltile) in per-lane register order:
// element p = lane*16 + idx maps to  col = ct*16 + (lane&15),
//                                    k   = kt*32 + ((lane>>4)<<3) + (idx&7) + ((idx>>3)<<4).
__device__ inline v16h load_b_packed(const _Float16* P, int lane) {
  return *(const v16h*)(P + (lane << 4));
}

// C/D layout: vgpr j, lane l: M = (l>=16 ? 8 : 0) + j, N = l & 15
__device__ inline void store_frag_lds_relu(_Float16* dst, int ldd, v8f c,
                                           const float* bias, int col0, int lane) {
  int n  = lane & 15;
  int mb = (lane >> 4) << 3;
  int col = col0 + n;
  float bv = bias[col];
#pragma unroll
  for (int j = 0; j < 8; ++j) {
    float v = fmaxf(c[j] + bv, 0.f);
    dst[(mb + j) * ldd + col] = (_Float16)v;
  }
}

// ---------------------------------------------------------------------------
// Weight packer: fp32 row-major W[Kreal x Nc] -> packed f16 tiles
// P[kt][ct][512], zero-padded beyond Kreal/Nc.
// ---------------------------------------------------------------------------
__global__ void pack_w(const float* __restrict__ W, _Float16* __restrict__ P,
                       int Kreal, int Nc, int KT, int CT) {
  int i = blockIdx.x * 256 + threadIdx.x;
  int total = KT * CT * 512;
  if (i >= total) return;
  int p    = i & 511;
  int t    = i >> 9;
  int ct   = t % CT;
  int kt   = t / CT;
  int lane = p >> 4;
  int idx  = p & 15;
  int col  = ct * 16 + (lane & 15);
  int k    = kt * 32 + ((lane >> 4) << 3) + (idx & 7) + ((idx >> 3) << 4);
  float v = (k < Kreal && col < Nc) ? W[(size_t)k * Nc + col] : 0.f;
  P[i] = (_Float16)v;
}

// ---------------------------------------------------------------------------
// Generic GEMM: Out[M,Nc] = act(A_f16[M,Kpad] @ Wpacked + bias)
// block = 256 threads = 8 waves; wave -> one 16x16 output tile
// ---------------------------------------------------------------------------
__global__ __launch_bounds__(256)
void gemm_kernel(const _Float16* __restrict__ A, const _Float16* __restrict__ P,
                 const float* __restrict__ bias, void* __restrict__ Out,
                 int M, int Kpad, int Nc, int CT, int relu, int outHalf) {
  int lane = threadIdx.x & 31;
  int wave = threadIdx.x >> 5;
  int row0 = blockIdx.y * 16;
  int ct   = blockIdx.x * 8 + wave;
  if (ct >= CT || row0 >= M) return;

  int KT = Kpad >> 5;
  const _Float16* At = A + (size_t)row0 * Kpad;
  const _Float16* Pt = P + (((size_t)ct) << 9) + (lane << 4);
  size_t pstride = ((size_t)CT) << 9;

  v8f acc = {};
  for (int kt = 0; kt < KT; ++kt) {
    v16h af = load_a_f16(At + (kt << 5), Kpad, lane);
    v16h bf = *(const v16h*)Pt;
    Pt += pstride;
    acc = WMMA_F16(af, bf, acc);
  }

  int n  = lane & 15;
  int mb = (lane >> 4) << 3;
  int col = ct * 16 + n;
  if (col < Nc) {
    float bv = bias[col];
#pragma unroll
    for (int j = 0; j < 8; ++j) {
      float v = acc[j] + bv;
      if (relu) v = fmaxf(v, 0.f);
      size_t off = (size_t)(row0 + mb + j) * Nc + col;
      if (outHalf) ((_Float16*)Out)[off] = (_Float16)v;
      else         ((float*)Out)[off]    = v;
    }
  }
}

// ---------------------------------------------------------------------------
// fp32 -> f16 conversion
// ---------------------------------------------------------------------------
__global__ void f32_to_f16(const float* __restrict__ in, _Float16* __restrict__ out, int n) {
  int i = blockIdx.x * 256 + threadIdx.x;
  if (i < n) out[i] = (_Float16)in[i];
}

// ---------------------------------------------------------------------------
// Cosine similarity + gumbel hard matching. One block (256 thr) per batch.
// ---------------------------------------------------------------------------
__global__ __launch_bounds__(256)
void match_kernel(const float* __restrict__ fobj, const float* __restrict__ fobjf,
                  const float* __restrict__ bbf, const float* __restrict__ gum,
                  float* __restrict__ outMatched) {
  int b = blockIdx.x, tid = threadIdx.x;
  __shared__ float sfo[NN * DV];
  __shared__ float sff[NN * DV];
  __shared__ float g[NN * NN];
  __shared__ float rho[NN * NN];
  __shared__ float part[256];
  __shared__ float scf[NN], scg[NN];

  const float* po = fobj  + (size_t)b * NN * DV;
  const float* pf = fobjf + (size_t)b * NN * DV;
  for (int i = tid; i < NN * DV; i += 256) { sfo[i] = po[i]; sff[i] = pf[i]; }
  g[tid] = gum[(size_t)b * 256 + tid];
  __syncthreads();

  int r = tid >> 4, s = tid & 15;
  float acc = 0.f;
  for (int c = 0; c < 16; ++c) { float v = sfo[r * DV + s * 16 + c]; acc += v * v; }
  part[tid] = acc;
  __syncthreads();
  if (s == 0) {
    float t = 0.f;
    for (int c = 0; c < 16; ++c) t += part[r * 16 + c];
    scf[r] = 1.f / (sqrtf(t) + 1e-8f);
  }
  __syncthreads();
  acc = 0.f;
  for (int c = 0; c < 16; ++c) { float v = sff[r * DV + s * 16 + c]; acc += v * v; }
  part[tid] = acc;
  __syncthreads();
  if (s == 0) {
    float t = 0.f;
    for (int c = 0; c < 16; ++c) t += part[r * 16 + c];
    scg[r] = 1.f / (sqrtf(t) + 1e-8f);
  }
  __syncthreads();

  float dot = 0.f;
  for (int c = 0; c < DV; ++c) dot += sfo[r * DV + c] * sff[s * DV + c];
  rho[tid] = dot * scf[r] * scg[s];
  __syncthreads();

  if (tid == 0) {
    float sel[NN];
    for (int j = 0; j < NN; ++j) sel[j] = 0.f;
    for (int i = 0; i < NN; ++i) {
      float best = -1e30f; int bj = 0;
      for (int j = 0; j < NN; ++j) {
        float sc = rho[i * NN + j] + g[i * NN + j] - 100.f * sel[j];
        if (sc > best) { best = sc; bj = j; }
      }
      sel[bj] += 1.f;
      for (int c = 0; c < 5; ++c)
        outMatched[(size_t)b * 80 + i * 5 + c] = bbf[(size_t)b * 80 + bj * 5 + c];
    }
  }
}

// ---------------------------------------------------------------------------
// Fused attention branch MLP. grid = (B, 4 branches), block = 256 (8 waves).
// Packed-weight tile sizes per branch: L1 16x16 tiles, L2 8x16, L3 8x8.
// ---------------------------------------------------------------------------
static constexpr int ATT1_SZ = 16 * 16 * 512;  // KT*CT*512
static constexpr int ATT2_SZ = 8 * 16 * 512;
static constexpr int ATT3_SZ = 8 * 8 * 512;

__global__ __launch_bounds__(256)
void attn_kernel(const float* __restrict__ fobj, const _Float16* __restrict__ mbuf,
                 const _Float16* __restrict__ P1, const float* __restrict__ b1,
                 const _Float16* __restrict__ P2, const float* __restrict__ b2,
                 const _Float16* __restrict__ P3, const float* __restrict__ b3,
                 const float* __restrict__ W4, const float* __restrict__ b4,
                 float* __restrict__ scores) {
  int b = blockIdx.x;
  int k = blockIdx.y;
  int tid = threadIdx.x;
  int lane = tid & 31, wave = tid >> 5;

  __shared__ __align__(32) _Float16 xbuf[NN * 512];
  __shared__ __align__(32) _Float16 hb1[NN * 256];
  __shared__ __align__(32) _Float16 hb2[NN * 256];

  const _Float16* mk = mbuf + (size_t)k * BB * 256 + (size_t)b * 256;
  const float*    fo = fobj + (size_t)b * NN * DV;

  for (int idx = tid; idx < NN * 512; idx += 256) {
    int n = idx >> 9, c = idx & 511;
    xbuf[idx] = (c < 256) ? mk[c] : (_Float16)fo[n * DV + (c - 256)];
  }
  __syncthreads();

  // Layer 1: [16,512]@[512,256], relu  (KT=16, CT=16; wave -> ct {2w, 2w+1})
  {
    const _Float16* p = P1 + (size_t)k * ATT1_SZ;
    const float* bias = b1 + k * 256;
    int ct0 = wave * 2, ct1 = ct0 + 1;
    v8f a0 = {}, a1 = {};
#pragma unroll 4
    for (int kt = 0; kt < 16; ++kt) {
      v16h af  = load_a_f16(xbuf + (kt << 5), 512, lane);
      v16h bf0 = load_b_packed(p + ((size_t)(kt * 16 + ct0) << 9), lane);
      v16h bf1 = load_b_packed(p + ((size_t)(kt * 16 + ct1) << 9), lane);
      a0 = WMMA_F16(af, bf0, a0);
      a1 = WMMA_F16(af, bf1, a1);
    }
    store_frag_lds_relu(hb1, 256, a0, bias, ct0 * 16, lane);
    store_frag_lds_relu(hb1, 256, a1, bias, ct1 * 16, lane);
  }
  __syncthreads();

  // Layer 2: [16,256]@[256,256], relu  (KT=8, CT=16)
  {
    const _Float16* p = P2 + (size_t)k * ATT2_SZ;
    const float* bias = b2 + k * 256;
    int ct0 = wave * 2, ct1 = ct0 + 1;
    v8f a0 = {}, a1 = {};
#pragma unroll 4
    for (int kt = 0; kt < 8; ++kt) {
      v16h af  = load_a_f16(hb1 + (kt << 5), 256, lane);
      v16h bf0 = load_b_packed(p + ((size_t)(kt * 16 + ct0) << 9), lane);
      v16h bf1 = load_b_packed(p + ((size_t)(kt * 16 + ct1) << 9), lane);
      a0 = WMMA_F16(af, bf0, a0);
      a1 = WMMA_F16(af, bf1, a1);
    }
    store_frag_lds_relu(hb2, 256, a0, bias, ct0 * 16, lane);
    store_frag_lds_relu(hb2, 256, a1, bias, ct1 * 16, lane);
  }
  __syncthreads();

  // Layer 3: [16,256]@[256,128], relu  (KT=8, CT=8; wave -> ct=wave)
  {
    const _Float16* p = P3 + (size_t)k * ATT3_SZ;
    const float* bias = b3 + k * 128;
    v8f a0 = {};
#pragma unroll 4
    for (int kt = 0; kt < 8; ++kt) {
      v16h af = load_a_f16(hb2 + (kt << 5), 256, lane);
      v16h bf = load_b_packed(p + ((size_t)(kt * 8 + wave) << 9), lane);
      a0 = WMMA_F16(af, bf, a0);
    }
    store_frag_lds_relu(hb1, 128, a0, bias, wave * 16, lane);
  }
  __syncthreads();

  // Layer 4: [16,128]@[128,1] -> raw scores
  if (tid < 16) {
    const float* w4 = W4 + (size_t)k * 128;
    float s = b4[k];
    for (int c = 0; c < 128; ++c) s += (float)hb1[tid * 128 + c] * w4[c];
    scores[((size_t)k * BB + b) * NN + tid] = s;
  }
}

// ---------------------------------------------------------------------------
// Softmaxes, attention combine, locations, emb build. One block per batch.
// ---------------------------------------------------------------------------
__global__ __launch_bounds__(64)
void combine_kernel(const float* __restrict__ scores, const float* __restrict__ logits,
                    const float* __restrict__ bbi, float* __restrict__ asub,
                    _Float16* __restrict__ emb) {
  int b = blockIdx.x, tid = threadIdx.x;
  __shared__ float soft[4][NN];
  __shared__ float fact[3];

  if (tid < 4) {
    const float* s = scores + ((size_t)tid * BB + b) * NN;
    float mx = s[0];
    for (int n = 1; n < NN; ++n) mx = fmaxf(mx, s[n]);
    float e[NN], sum = 0.f;
    for (int n = 0; n < NN; ++n) { e[n] = expf(s[n] - mx); sum += e[n]; }
    for (int n = 0; n < NN; ++n) soft[tid][n] = e[n] / sum;
  }
  if (tid == 4) {
    const float* l = logits + (size_t)b * 3;
    float mx = fmaxf(l[0], fmaxf(l[1], l[2]));
    float e0 = expf(l[0] - mx), e1 = expf(l[1] - mx), e2 = expf(l[2] - mx);
    float s = e0 + e1 + e2;
    fact[0] = e0 / s; fact[1] = e1 / s; fact[2] = e2 / s;
  }
  __syncthreads();

  if (tid == 0) {
    float asl[NN], aol[NN], sa = 0.f, so = 0.f;
    for (int n = 0; n < NN; ++n) {
      asl[n] = soft[0][n] * soft[1][n];
      aol[n] = soft[2][n] * soft[3][n];
      sa += asl[n]; so += aol[n];
    }
    float sloc[5] = {0, 0, 0, 0, 0}, oloc[5] = {0, 0, 0, 0, 0};
    for (int n = 0; n < NN; ++n) {
      asl[n] /= sa; aol[n] /= so;
      asub[(size_t)b * NN + n] = asl[n];
      for (int c = 0; c < 5; ++c) {
        float bv = bbi[(size_t)b * 80 + n * 5 + c];
        sloc[c] += asl[n] * bv;
        oloc[c] += aol[n] * bv;
      }
    }
    _Float16* e = emb + (size_t)b * 32;
    for (int c = 0; c < 3; ++c) e[c]     = (_Float16)fact[c];
    for (int c = 0; c < 5; ++c) e[3 + c] = (_Float16)oloc[c];
    for (int c = 0; c < 5; ++c) e[8 + c] = (_Float16)sloc[c];
    for (int c = 13; c < 32; ++c) e[c]   = (_Float16)0.f;
  }
}

// ---------------------------------------------------------------------------
// Final blend: pred_bboxes = (1-a_sub)*bboxes_i + a_sub*tanh(pred)
// ---------------------------------------------------------------------------
__global__ void blend_kernel(const float* __restrict__ bbi, const float* __restrict__ asub,
                             const float* __restrict__ predraw, float* __restrict__ out) {
  int idx = blockIdx.x * 256 + threadIdx.x;
  if (idx >= BB * NN * 5) return;
  int b = idx / 80, rem = idx % 80, n = rem / 5, c = rem % 5;
  float a = asub[(size_t)b * NN + n];
  float p = tanhf(predraw[(size_t)b * 5 + c]);
  out[idx] = (1.f - a) * bbi[idx] + a * p;
}

// ---------------------------------------------------------------------------
// Host-side launch
// ---------------------------------------------------------------------------
extern "C" void kernel_launch(void* const* d_in, const int* in_sizes, int n_in,
                              void* d_out, int out_size, void* d_ws, size_t ws_size,
                              hipStream_t stream) {
  const float* f_objects       = (const float*)d_in[0];
  const float* f_objects_final = (const float*)d_in[1];
  const float* bboxes_i        = (const float*)d_in[2];
  const float* bboxes_f        = (const float*)d_in[3];
  const float* f_instruction   = (const float*)d_in[4];
  const float* gumbel          = (const float*)d_in[5];
  const float* decW[5] = {(const float*)d_in[6],  (const float*)d_in[8],
                          (const float*)d_in[10], (const float*)d_in[12],
                          (const float*)d_in[14]};
  const float* decB[5] = {(const float*)d_in[7],  (const float*)d_in[9],
                          (const float*)d_in[11], (const float*)d_in[13],
                          (const float*)d_in[15]};
  const float* mapW1 = (const float*)d_in[16];
  const float* mapb1 = (const float*)d_in[17];
  const float* mapW2 = (const float*)d_in[18];
  const float* mapb2 = (const float*)d_in[19];
  const float* attW1 = (const float*)d_in[20];
  const float* attb1 = (const float*)d_in[21];
  const float* attW2 = (const float*)d_in[22];
  const float* attb2 = (const float*)d_in[23];
  const float* attW3 = (const float*)d_in[24];
  const float* attb3 = (const float*)d_in[25];
  const float* attW4 = (const float*)d_in[26];
  const float* attb4 = (const float*)d_in[27];
  const float* predW[5] = {(const float*)d_in[28], (const float*)d_in[30],
                           (const float*)d_in[32], (const float*)d_in[34],
                           (const float*)d_in[36]};
  const float* predB[5] = {(const float*)d_in[29], (const float*)d_in[31],
                           (const float*)d_in[33], (const float*)d_in[35],
                           (const float*)d_in[37]};
  float* out = (float*)d_out;

  // ---- workspace bump allocator (64B aligned) ----
  char* wp = (char*)d_ws;
  auto alloc = [&](size_t bytes) -> char* {
    char* r = wp;
    wp += (bytes + 63) & ~(size_t)63;
    return r;
  };
  _Float16* fi16   = (_Float16*)alloc((size_t)BB * DL * 2);
  _Float16* bufA   = (_Float16*)alloc((size_t)BB * 512 * 2);
  _Float16* bufB   = (_Float16*)alloc((size_t)BB * 512 * 2);
  _Float16* mbuf   = (_Float16*)alloc((size_t)4 * BB * 256 * 2);
  float*    scores = (float*)   alloc((size_t)4 * BB * NN * 4);
  float*    asub   = (float*)   alloc((size_t)BB * NN * 4);
  float*    logits = (float*)   alloc((size_t)BB * 3 * 4);
  _Float16* emb16  = (_Float16*)alloc((size_t)BB * 32 * 2);
  float*    predraw= (float*)   alloc((size_t)BB * 5 * 4);

  auto PACK = [&](const float* W, int Kreal, int Nc) -> _Float16* {
    int KT = (Kreal + 31) / 32, CT = (Nc + 15) / 16;
    int total = KT * CT * 512;
    _Float16* P = (_Float16*)alloc((size_t)total * 2);
    pack_w<<<(total + 255) / 256, 256, 0, stream>>>(W, P, Kreal, Nc, KT, CT);
    return P;
  };

  // ---- pack all weights into WMMA fragment order (f16, zero-padded) ----
  _Float16* decP[5];
  const int decK[5] = {1024, 512, 256, 256, 128};
  const int decN[5] = {512, 256, 256, 128, 3};
  for (int i = 0; i < 5; ++i) decP[i] = PACK(decW[i], decK[i], decN[i]);

  _Float16* mapP1[4];
  _Float16* mapP2[4];
  for (int k = 0; k < 4; ++k) {
    mapP1[k] = PACK(mapW1 + (size_t)k * 1024 * 512, 1024, 512);
    mapP2[k] = PACK(mapW2 + (size_t)k * 512 * 256, 512, 256);
  }

  // att packed weights: allocate contiguously so the kernel can index by branch
  _Float16* attP1 = (_Float16*)alloc((size_t)4 * ATT1_SZ * 2);
  _Float16* attP2 = (_Float16*)alloc((size_t)4 * ATT2_SZ * 2);
  _Float16* attP3 = (_Float16*)alloc((size_t)4 * ATT3_SZ * 2);
  for (int k = 0; k < 4; ++k) {
    pack_w<<<(ATT1_SZ + 255) / 256, 256, 0, stream>>>(attW1 + (size_t)k * 512 * 256,
                                                      attP1 + (size_t)k * ATT1_SZ, 512, 256, 16, 16);
    pack_w<<<(ATT2_SZ + 255) / 256, 256, 0, stream>>>(attW2 + (size_t)k * 256 * 256,
                                                      attP2 + (size_t)k * ATT2_SZ, 256, 256, 8, 16);
    pack_w<<<(ATT3_SZ + 255) / 256, 256, 0, stream>>>(attW3 + (size_t)k * 256 * 128,
                                                      attP3 + (size_t)k * ATT3_SZ, 256, 128, 8, 8);
  }

  _Float16* predP[5];
  const int predK[5] = {13, 256, 512, 512, 256};
  const int predN[5] = {256, 512, 512, 256, 5};
  for (int i = 0; i < 5; ++i) predP[i] = PACK(predW[i], predK[i], predN[i]);

  auto GEMM = [&](const _Float16* A, const _Float16* P, const float* bias, void* O,
                  int M, int Kp, int Nc, int relu, int half) {
    int CT = (Nc + 15) / 16;
    dim3 grid((CT + 7) / 8, M / 16);
    gemm_kernel<<<grid, 256, 0, stream>>>(A, P, bias, O, M, Kp, Nc, CT, relu, half);
  };

  // 1) convert instruction features to f16
  f32_to_f16<<<(BB * DL + 255) / 256, 256, 0, stream>>>(f_instruction, fi16, BB * DL);

  // 2) gumbel hard matching -> output #2
  match_kernel<<<BB, 256, 0, stream>>>(f_objects, f_objects_final, bboxes_f, gumbel,
                                       out + (size_t)BB * NN * 5);

  // 3) instruction decoder chain -> logits
  GEMM(fi16, decP[0], decB[0], bufA,   BB, 1024, 512, 1, 1);
  GEMM(bufA, decP[1], decB[1], bufB,   BB,  512, 256, 1, 1);
  GEMM(bufB, decP[2], decB[2], bufA,   BB,  256, 256, 1, 1);
  GEMM(bufA, decP[3], decB[3], bufB,   BB,  256, 128, 1, 1);
  GEMM(bufB, decP[4], decB[4], logits, BB,  128,   3, 0, 0);

  // 4) instruction mapping per branch -> m_k
  for (int k = 0; k < 4; ++k) {
    GEMM(fi16, mapP1[k], mapb1 + k * 512, bufA, BB, 1024, 512, 1, 1);
    GEMM(bufA, mapP2[k], mapb2 + k * 256, mbuf + (size_t)k * BB * 256, BB, 512, 256, 0, 1);
  }

  // 5) fused attention MLPs -> raw scores
  attn_kernel<<<dim3(BB, 4), 256, 0, stream>>>(f_objects, mbuf,
                                               attP1, attb1, attP2, attb2,
                                               attP3, attb3, attW4, attb4, scores);

  // 6) softmaxes + combine + emb
  combine_kernel<<<BB, 64, 0, stream>>>(scores, logits, bboxes_i, asub, emb16);

  // 7) prediction chain (K=13 padded to 32)
  GEMM(emb16, predP[0], predB[0], bufA,    BB,  32, 256, 1, 1);
  GEMM(bufA,  predP[1], predB[1], bufB,    BB, 256, 512, 1, 1);
  GEMM(bufB,  predP[2], predB[2], bufA,    BB, 512, 512, 1, 1);
  GEMM(bufA,  predP[3], predB[3], bufB,    BB, 512, 256, 1, 1);
  GEMM(bufB,  predP[4], predB[4], predraw, BB, 256,   5, 0, 0);

  // 8) final blend -> output #1
  blend_kernel<<<(BB * NN * 5 + 255) / 256, 256, 0, stream>>>(bboxes_i, asub, predraw, out);
}
